// CostVolume_14877766714072
// MI455X (gfx1250) — matrix operand order repaired and checked by reading, not compiled, not run on hardware
//
#include <hip/hip_runtime.h>

// Cost volume via banded-Gram WMMA on gfx1250 (wave32).
// out[b,h,w,di*9+dj] = leaky( mean_c( prv[b,h,w,c] * nxt[b,h+di-4,w+dj-4,c] ) )

#define B_ 8
#define H_ 128
#define W_ 256
#define C_ 192
#define R_ 4
#define D_ 9           // 2R+1
#define ND_ 81         // D*D
#define WTILE 32       // pixels per workgroup along W
#define HALO 40        // WTILE + 2R
#define CPAD 200       // C padded (ushort units): 400B pixel stride -> gcd(100,64)=4 bank spread
#define BLOCK 256      // 8 waves
#define NWAVES 8
#define NJOBS 36       // 2 M-tiles * 9 di * 2 N-tiles

typedef __attribute__((ext_vector_type(16))) __bf16 v16bf;
typedef __attribute__((ext_vector_type(2)))  __bf16 v2bf;
typedef __attribute__((ext_vector_type(8)))  float  v8f;
typedef __attribute__((ext_vector_type(4)))  float  f32x4;

union Frag { uint4 q[2]; v16bf v; };

static __device__ __forceinline__ unsigned int pack_bf16(float a, float b) {
    union { v2bf v; unsigned int u; } cv;
    cv.v.x = (__bf16)a;            // native cvt path
    cv.v.y = (__bf16)b;
    return cv.u;
}

__global__ __launch_bounds__(BLOCK)
void CostVolume_14877766714072_kernel(const float* __restrict__ prv,
                                      const float* __restrict__ nxt,
                                      float* __restrict__ out,
                                      float* __restrict__ ws) {
    __shared__ __align__(16) unsigned short s_nxt[D_ * HALO * CPAD]; // 144000 B
    __shared__ __align__(16) unsigned short s_prv[WTILE * CPAD];     //  12800 B

    const int wbase = blockIdx.x * WTILE;
    const int h     = blockIdx.y;
    const int bb    = blockIdx.z;
    const int tid   = threadIdx.x;

    // ---- Stage nxt halo (9 rows x 40 px x 192 ch) f32 -> bf16 into LDS ----
    // nxt rows are reused ~11x across neighboring WGs: regular (L2-resident) loads.
    for (int t = tid; t < D_ * HALO * (C_ / 4); t += BLOCK) {
        const int chunk = t % (C_ / 4);
        const int pr    = t / (C_ / 4);
        const int pix   = pr % HALO;
        const int row   = pr / HALO;
        const int hh    = h + row - R_;
        const int ww    = wbase + pix - R_;
        f32x4 v = {0.f, 0.f, 0.f, 0.f};
        if (hh >= 0 && hh < H_ && ww >= 0 && ww < W_)
            v = *(const f32x4*)&nxt[(((size_t)bb * H_ + hh) * W_ + ww) * C_ + chunk * 4];
        uint2 p;
        p.x = pack_bf16(v.x, v.y);
        p.y = pack_bf16(v.z, v.w);
        *(uint2*)&s_nxt[(row * HALO + pix) * CPAD + chunk * 4] = p;
    }
    // ---- Stage prv tile (32 px x 192 ch): touched exactly once -> non-temporal ----
    for (int t = tid; t < WTILE * (C_ / 4); t += BLOCK) {
        const int chunk = t % (C_ / 4);
        const int pix   = t / (C_ / 4);
        f32x4 v = __builtin_nontemporal_load(
            (const f32x4*)&prv[(((size_t)bb * H_ + h) * W_ + (wbase + pix)) * C_ + chunk * 4]);
        uint2 p;
        p.x = pack_bf16(v.x, v.y);
        p.y = pack_bf16(v.z, v.w);
        *(uint2*)&s_prv[pix * CPAD + chunk * 4] = p;
    }
    __syncthreads();

    const int lane = tid & 31;
    const int wave = tid >> 5;
    const int lo16 = lane & 15;
    const int hi   = lane >> 4;

    // Linear output index: idx(r) = P + 1296*mi + 9*di + 16*ni + lo16 + 640*hi + 80*r
    const size_t P  = ((size_t)(bb * H_ + h) * W_ + wbase) * ND_;
    float* const dump = ws + (tid & 63);          // dump target for out-of-band lanes

    const int ni   = wave & 1;                    // N-tile: fixed per wave (stride 8, even)
    int di = wave >> 1;                           // 0..3
    int mi = 0;
    const int pl   = lo16 + 640 * hi + 16 * ni;   // per-lane constant part of idx
    const int obas = 16 * ni + lo16 - 8 * hi - 4; // o(r) = obas - r ; valid iff o in [-4,4]

    for (int j = wave; j < NJOBS; j += NWAVES) {
        // B column pixel in halo coords (clamped; out-of-band cols discarded below)
        int bcol = 16 * mi + 16 * ni + lo16;
        bcol = bcol > (HALO - 1) ? (HALO - 1) : bcol;
        const unsigned short* bbase = &s_nxt[(di * HALO + bcol) * CPAD];
        const unsigned short* abase = &s_prv[(16 * mi + lo16) * CPAD];

        v8f acc = {};
        #pragma unroll
        for (int ks = 0; ks < C_ / 32; ++ks) {
            const int c0 = ks * 32;
            Frag A, Bf;
            // A 16x32 bf16: lane(lo16)=row M; lanes 0-15 carry K {0..7,16..23}, 16-31 carry {8..15,24..31}
            A.q[0]  = *(const uint4*)(abase + c0 + 8 * hi);
            A.q[1]  = *(const uint4*)(abase + c0 + 8 * hi + 16);
            // B 32x16 bf16: lane(lo16)=col N; lanes 0-15 carry K {0..7,16..23}, 16-31 carry {8..15,24..31}
            Bf.q[0] = *(const uint4*)(bbase + c0 + 8 * hi);
            Bf.q[1] = *(const uint4*)(bbase + c0 + 8 * hi + 16);
            acc = __builtin_amdgcn_wmma_f32_16x16x32_bf16(
                      false, A.v, false, Bf.v, (short)0, acc, false, false);
        }

        // One base address per job; 8 stores at immediate offsets 320*r.
        float* const obptr = out + (P + 1296 * mi + 9 * di + pl);
        #pragma unroll
        for (int r = 0; r < 8; ++r) {
            const bool valid = (unsigned)(obas - r + R_) <= (unsigned)(2 * R_);
            float v = acc[r] * (1.0f / (float)C_);
            v = (v >= 0.0f) ? v : 0.1f * v;
            float* dst = (valid ? obptr : dump) + 80 * r;
            __builtin_nontemporal_store(v, dst);   // write-once stream, keep L2 for nxt
        }

        di += 4;                                   // incremental (di,mi) decode, no division
        if (di >= D_) { di -= D_; mi += 1; }
    }
}

extern "C" void kernel_launch(void* const* d_in, const int* in_sizes, int n_in,
                              void* d_out, int out_size, void* d_ws, size_t ws_size,
                              hipStream_t stream) {
    const float* prv = (const float*)d_in[0];
    const float* nxt = (const float*)d_in[1];
    float* out = (float*)d_out;
    float* ws  = (float*)d_ws;
    dim3 grid(W_ / WTILE, H_, B_);   // (8, 128, 8)
    dim3 block(BLOCK);
    CostVolume_14877766714072_kernel<<<grid, block, 0, stream>>>(prv, nxt, out, ws);
}
